// CausalMHA_36996848287912
// MI455X (gfx1250) — compile-verified, matrix operand order checked by reading, not run on hardware
//
#include <hip/hip_runtime.h>
#include <hip/hip_bf16.h>

// Causal MHA + RoPE for MI455X (gfx1250, wave32).
// All matmuls on v_wmma_f32_16x16x32_bf16 (fp32 accumulate).
// GEMMs use double-buffered LDS tiles, staged with async global->LDS copies
// (ASYNCcnt path) when the toolchain exposes the builtins.

#define B_  2
#define L_  2048
#define DM  1024
#define NH  16
#define HD  64
#define E3  3072
#define BL  (B_ * L_)

typedef __attribute__((ext_vector_type(16))) __bf16        v16bf;
typedef __attribute__((ext_vector_type(8)))  float         v8f;
typedef __attribute__((ext_vector_type(4)))  unsigned int  v4u;
typedef __attribute__((ext_vector_type(4)))  int           v4i;

union FragU { v16bf v; v4u u[2]; };

// ---- gfx1250 async global->LDS copy (guarded; fallback = load + ds_store) ----
#if defined(__has_builtin)
#  if __has_builtin(__builtin_amdgcn_global_load_async_to_lds_b128)
#    define HAVE_ASYNC_LDS 1
#  endif
#  if __has_builtin(__builtin_amdgcn_s_wait_asynccnt)
#    define HAVE_WAIT_ASYNC 1
#  endif
#endif

#if defined(HAVE_ASYNC_LDS)
#  define ASYNC_CP_B128(l, g)                                              \
    __builtin_amdgcn_global_load_async_to_lds_b128(                        \
        (__attribute__((address_space(1))) v4i*)(g),                       \
        (__attribute__((address_space(3))) v4i*)(l), 0, 0)
#  if defined(HAVE_WAIT_ASYNC)
#    define WAIT_ASYNC() __builtin_amdgcn_s_wait_asynccnt(0)
#  else
#    define WAIT_ASYNC() asm volatile("s_wait_asynccnt 0x0" ::: "memory")
#  endif
#else
#  define ASYNC_CP_B128(l, g) (*(v4u*)(l) = *(const v4u*)(g))
#  define WAIT_ASYNC()
#endif

__device__ __forceinline__ unsigned short f2bf(float f) {
  unsigned int u = __float_as_uint(f);
  u += 0x7FFFu + ((u >> 16) & 1u);          // round-to-nearest-even
  return (unsigned short)(u >> 16);
}

__device__ __forceinline__ v8f wmma_bf16(v16bf a, v16bf b, v8f c) {
  // (neg_a, A, neg_b, B, c_mod, C, reuse_a, reuse_b)
  return __builtin_amdgcn_wmma_f32_16x16x32_bf16(false, a, false, b, (short)0, c,
                                                 false, false);
}

// 16-bit A/B operand fragment (16x32 MxK tile, row-major source).
// lane 0-15: row=lane,  K = {0..7, 16..23};  lane 16-31: row=lane-16, K = {8..15, 24..31}
__device__ __forceinline__ v16bf load_frag_g(const unsigned short* base, int ld,
                                             int k0, int lane) {
  const unsigned short* p =
      base + (size_t)(lane & 15) * (size_t)ld + k0 + ((lane >> 4) << 3);
  FragU f;
  f.u[0] = *(const v4u*)(p);
  f.u[1] = *(const v4u*)(p + 16);
  return f.v;
}

__device__ __forceinline__ v16bf load_frag_lds(const unsigned short* base, int ld,
                                               int lane) {
  const unsigned short* p = base + (lane & 15) * ld + ((lane >> 4) << 3);
  FragU f;
  f.u[0] = *(const v4u*)(p);
  f.u[1] = *(const v4u*)(p + 16);
  return f.v;
}

// ---------------------------------------------------------------- convert f32->bf16
__global__ void cvt_f32_bf16(const float* __restrict__ src,
                             unsigned short* __restrict__ dst, int n) {
  int i = blockIdx.x * blockDim.x + threadIdx.x;
  if (i < n) dst[i] = f2bf(src[i]);
}

// ---------------------------------------------------------------- bf16 WMMA GEMM
// C[M,N] (f32) = A[M,K](bf16 row-major) * B[N,K](bf16 row-major)^T
// WG = 8 waves = 128x256 tile; each wave 64x64 (4x4 wmma accumulators).
// K-steps of 32 staged through double-buffered LDS (rows padded to 40 elems ->
// 80B row stride = 20 banks: the 16-lane b128 fragment reads hit 16 distinct banks).
#define BM   128
#define BN   256
#define BK   32
#define LDT  40                      // padded row length (elements)
#define A_ELE (BM * LDT)             // 5120
#define B_ELE (BN * LDT)             // 10240

__global__ void __launch_bounds__(256)
gemm_bf16_f32(const unsigned short* __restrict__ A,
              const unsigned short* __restrict__ Bm,
              float* __restrict__ C,
              int K, int lda, int ldb, int ldc) {
  __shared__ __align__(16) unsigned short smem[2 * (A_ELE + B_ELE)];

  const int tid  = threadIdx.x;
  const int lane = tid & 31;
  const int wid  = tid >> 5;
  const int mblk = blockIdx.y * BM;
  const int nblk = blockIdx.x * BN;
  const int m0   = (wid & 1) * 64;
  const int n0   = (wid >> 1) * 64;

  auto stage = [&](int buf, int k0) {
    unsigned short* sA = smem + buf * (A_ELE + B_ELE);
    unsigned short* sB = sA + A_ELE;
    // A tile: 128 rows x 32 cols = 512 x 16B chunks, 2 per thread
#pragma unroll
    for (int i = 0; i < 2; ++i) {
      int c = tid + i * 256;
      int row = c >> 2, cc = (c & 3) * 8;
      const unsigned short* g = A + (size_t)(mblk + row) * (size_t)lda + k0 + cc;
      ASYNC_CP_B128(&sA[row * LDT + cc], g);
    }
    // B tile: 256 rows x 32 cols = 1024 x 16B chunks, 4 per thread
#pragma unroll
    for (int i = 0; i < 4; ++i) {
      int c = tid + i * 256;
      int row = c >> 2, cc = (c & 3) * 8;
      const unsigned short* g = Bm + (size_t)(nblk + row) * (size_t)ldb + k0 + cc;
      ASYNC_CP_B128(&sB[row * LDT + cc], g);
    }
  };

  v8f acc[4][4];
  const v8f zf = {0.f, 0.f, 0.f, 0.f, 0.f, 0.f, 0.f, 0.f};
#pragma unroll
  for (int i = 0; i < 4; ++i)
#pragma unroll
    for (int j = 0; j < 4; ++j) acc[i][j] = zf;

  const int nk = K / BK;
  stage(0, 0);

  for (int kt = 0; kt < nk; ++kt) {
    WAIT_ASYNC();            // own async copies of buffer kt complete
    __syncthreads();         // => everyone's copies complete; prior reads drained

    if (kt + 1 < nk) stage((kt + 1) & 1, (kt + 1) * BK);  // overlap with compute

    const unsigned short* sA = smem + (kt & 1) * (A_ELE + B_ELE);
    const unsigned short* sB = sA + A_ELE;

    v16bf a[4];
#pragma unroll
    for (int i = 0; i < 4; ++i)
      a[i] = load_frag_lds(sA + (m0 + i * 16) * LDT, LDT, lane);
#pragma unroll
    for (int j = 0; j < 4; ++j) {
      v16bf b = load_frag_lds(sB + (n0 + j * 16) * LDT, LDT, lane);
#pragma unroll
      for (int i = 0; i < 4; ++i)
        acc[i][j] = wmma_bf16(a[i], b, acc[i][j]);
    }
  }

  // C/D layout: VGPR r -> row (+8 for upper half-wave), lane&15 -> col
  const int half8 = (lane >> 4) << 3;
  const int cl    = lane & 15;
#pragma unroll
  for (int i = 0; i < 4; ++i)
#pragma unroll
    for (int j = 0; j < 4; ++j) {
      float* cp = C + (size_t)(mblk + m0 + i * 16 + half8) * (size_t)ldc +
                  nblk + n0 + j * 16 + cl;
#pragma unroll
      for (int r = 0; r < 8; ++r) cp[(size_t)r * (size_t)ldc] = acc[i][j][r];
    }
}

// ---------------------------------------------------------------- RoPE + repack
// qkv f32 [B,L,3,H,HD] -> Q,K bf16 [B*H, L, HD] (scale 1/8 folded into Q),
//                          V  bf16 transposed [B*H, HD, L]
__global__ void rope_pack(const float* __restrict__ qkv,
                          unsigned short* __restrict__ Qb,
                          unsigned short* __restrict__ Kb,
                          unsigned short* __restrict__ Vt) {
  int idx = blockIdx.x * blockDim.x + threadIdx.x;   // B*L*H*32 threads
  int j = idx & 31;
  int h = (idx >> 5) & (NH - 1);
  int l = (idx >> 9) & (L_ - 1);
  int b = idx >> 20;

  const float* row  = qkv + (size_t)(b * L_ + l) * E3;
  const float* qrow = row;
  const float* krow = row + DM;
  const float* vrow = row + 2 * DM;

  float inv = __powf(10000.0f, -(float)j * (1.0f / 32.0f));
  float ang = (float)l * inv;
  float c = __cosf(ang), s = __sinf(ang);

  int d1 = h * HD + j, d2 = d1 + 32;
  float q1 = qrow[d1], q2 = qrow[d2];
  float k1 = krow[d1], k2 = krow[d2];

  const float qs = 0.125f;  // 1/sqrt(HD) folded into Q
  size_t bh = (size_t)(b * NH + h);
  size_t qo = (bh * L_ + l) * HD;
  Qb[qo + j]      = f2bf((q1 * c - q2 * s) * qs);
  Qb[qo + j + 32] = f2bf((q2 * c + q1 * s) * qs);
  Kb[qo + j]      = f2bf(k1 * c - k2 * s);
  Kb[qo + j + 32] = f2bf(k2 * c + k1 * s);

  size_t vo = bh * HD * L_ + l;
  Vt[vo + (size_t)j * L_]        = f2bf(vrow[d1]);
  Vt[vo + (size_t)(j + 32) * L_] = f2bf(vrow[d2]);
}

// ---------------------------------------------------------------- flash attention
// One wave per (b*h, 16-query tile). Online softmax, key blocks of 32.
__global__ void __launch_bounds__(256)
attn_kernel(const unsigned short* __restrict__ Q,
            const unsigned short* __restrict__ K,
            const unsigned short* __restrict__ Vt,
            unsigned short* __restrict__ hb) {
  __shared__ __align__(16) unsigned short sP[8 * 16 * 32];  // per-wave P tile

  const int lane = threadIdx.x & 31;
  const int wid  = threadIdx.x >> 5;
  const int task = blockIdx.x * 8 + wid;        // B*H*(L/16) = 4096 tasks
  const int qt   = task & (L_ / 16 - 1);
  const int bh   = task >> 7;
  const int q0   = qt * 16;

  const unsigned short* Qp = Q  + (size_t)bh * L_ * HD;
  const unsigned short* Kp = K  + (size_t)bh * L_ * HD;
  const unsigned short* Vp = Vt + (size_t)bh * HD * L_;
  unsigned short* sp = sP + wid * (16 * 32);

  v16bf aq0 = load_frag_g(Qp + (size_t)q0 * HD, HD, 0,  lane);
  v16bf aq1 = load_frag_g(Qp + (size_t)q0 * HD, HD, 32, lane);

  const v8f zf = {0.f, 0.f, 0.f, 0.f, 0.f, 0.f, 0.f, 0.f};
  v8f acc[4];
#pragma unroll
  for (int t = 0; t < 4; ++t) acc[t] = zf;

  float mrow[8], lrow[8];
#pragma unroll
  for (int r = 0; r < 8; ++r) { mrow[r] = -__builtin_inff(); lrow[r] = 0.f; }

  const int half8 = (lane >> 4) << 3;
  const int cl    = lane & 15;
  const int nblk  = (q0 + 16 + 31) >> 5;

  for (int kb = 0; kb < nblk; ++kb) {
    const int kk = kb * 32;

    v8f s0 = zf, s1 = zf;
    {
      v16bf bk = load_frag_g(Kp + (size_t)kk * HD, HD, 0, lane);
      s0 = wmma_bf16(aq0, bk, s0);
      bk = load_frag_g(Kp + (size_t)kk * HD, HD, 32, lane);
      s0 = wmma_bf16(aq1, bk, s0);
      bk = load_frag_g(Kp + (size_t)(kk + 16) * HD, HD, 0, lane);
      s1 = wmma_bf16(aq0, bk, s1);
      bk = load_frag_g(Kp + (size_t)(kk + 16) * HD, HD, 32, lane);
      s1 = wmma_bf16(aq1, bk, s1);
    }

    float alpha[8];
#pragma unroll
    for (int r = 0; r < 8; ++r) {
      const int qrow = q0 + r + half8;
      if (kk + cl > qrow)      s0[r] = -__builtin_inff();
      if (kk + 16 + cl > qrow) s1[r] = -__builtin_inff();

      float v = fmaxf(s0[r], s1[r]);
#pragma unroll
      for (int m = 8; m; m >>= 1) v = fmaxf(v, __shfl_xor(v, m, 16));
      float mnew = fmaxf(mrow[r], v);
      float a    = __expf(mrow[r] - mnew);
      float p0   = __expf(s0[r] - mnew);
      float p1   = __expf(s1[r] - mnew);
      float rs   = p0 + p1;
#pragma unroll
      for (int m = 8; m; m >>= 1) rs += __shfl_xor(rs, m, 16);
      lrow[r]  = lrow[r] * a + rs;
      mrow[r]  = mnew;
      alpha[r] = a;
      s0[r] = p0;
      s1[r] = p1;
    }

#pragma unroll
    for (int t = 0; t < 4; ++t)
#pragma unroll
      for (int r = 0; r < 8; ++r) acc[t][r] *= alpha[r];

    // P (C-layout) -> LDS row-major 16x32 bf16, reload as A-operand fragment
#pragma unroll
    for (int r = 0; r < 8; ++r) {
      const int row = r + half8;
      sp[row * 32 + cl]      = f2bf(s0[r]);
      sp[row * 32 + 16 + cl] = f2bf(s1[r]);
    }
    v16bf pa = load_frag_lds(sp, 32, lane);

#pragma unroll
    for (int t = 0; t < 4; ++t) {
      v16bf bv = load_frag_g(Vp + (size_t)(t * 16) * L_, L_, kk, lane);
      acc[t] = wmma_bf16(pa, bv, acc[t]);
    }
  }

  const int b = bh >> 4;
  const int h = bh & 15;
#pragma unroll
  for (int t = 0; t < 4; ++t)
#pragma unroll
    for (int r = 0; r < 8; ++r) {
      const int qrow = q0 + r + half8;
      const float o = acc[t][r] / lrow[r];
      hb[(size_t)(b * L_ + qrow) * DM + h * HD + t * 16 + cl] = f2bf(o);
    }
}

// ---------------------------------------------------------------- launch
extern "C" void kernel_launch(void* const* d_in, const int* in_sizes, int n_in,
                              void* d_out, int out_size, void* d_ws, size_t ws_size,
                              hipStream_t stream) {
  (void)in_sizes; (void)n_in; (void)out_size; (void)ws_size;
  const float* x    = (const float*)d_in[0];
  const float* Wqkv = (const float*)d_in[1];
  const float* Wout = (const float*)d_in[2];
  float* out = (float*)d_out;

  char* ws = (char*)d_ws;
  unsigned short* xb    = (unsigned short*)(ws);                           //  8 MB
  unsigned short* Wqkvb = (unsigned short*)(ws + (size_t)8  * (1u << 20)); //  6 MB
  unsigned short* Woutb = (unsigned short*)(ws + (size_t)14 * (1u << 20)); //  2 MB
  float*          qkv   = (float*)        (ws + (size_t)16 * (1u << 20)); // 48 MB
  unsigned short* Qb    = (unsigned short*)(ws + (size_t)64 * (1u << 20)); //  8 MB
  unsigned short* Kb    = (unsigned short*)(ws + (size_t)72 * (1u << 20)); //  8 MB
  unsigned short* Vt    = (unsigned short*)(ws + (size_t)80 * (1u << 20)); //  8 MB
  unsigned short* hb    = (unsigned short*)(ws + (size_t)88 * (1u << 20)); //  8 MB

  const int nx = BL * DM, nqkv = E3 * DM, nout = DM * DM;
  cvt_f32_bf16<<<(nx   + 255) / 256, 256, 0, stream>>>(x,    xb,    nx);
  cvt_f32_bf16<<<(nqkv + 255) / 256, 256, 0, stream>>>(Wqkv, Wqkvb, nqkv);
  cvt_f32_bf16<<<(nout + 255) / 256, 256, 0, stream>>>(Wout, Woutb, nout);

  // qkv = x * Wqkv^T   (M=4096, N=3072, K=1024)
  gemm_bf16_f32<<<dim3(E3 / BN, BL / BM), 256, 0, stream>>>(
      xb, Wqkvb, qkv, DM, DM, DM, E3);

  rope_pack<<<(B_ * L_ * NH * 32) / 256, 256, 0, stream>>>(qkv, Qb, Kb, Vt);

  attn_kernel<<<(B_ * NH * (L_ / 16)) / 8, 256, 0, stream>>>(Qb, Kb, Vt, hb);

  // out = h * Wout^T   (M=4096, N=1024, K=1024)
  gemm_bf16_f32<<<dim3(DM / BN, BL / BM), 256, 0, stream>>>(
      hb, Woutb, out, DM, DM, DM, DM);
}
